// MoE_61649960566989
// MI455X (gfx1250) — matrix operand order repaired and checked by reading, not compiled, not run on hardware
//
#include <hip/hip_runtime.h>

// ---------------------------------------------------------------------------
// Types for WMMA
// ---------------------------------------------------------------------------
typedef __attribute__((ext_vector_type(16))) __bf16 v16bf;
typedef __attribute__((ext_vector_type(8)))  float  v8f;

__device__ __forceinline__ unsigned short f2bf(float f) {
    union { float f; unsigned u; } v; v.f = f;
    unsigned r = v.u + 0x7FFFu + ((v.u >> 16) & 1u);   // round-to-nearest-even
    return (unsigned short)(r >> 16);
}

// CDNA5 async global->LDS copy (ASYNCcnt-tracked, no VGPR data path).
// GVS mode: mem_addr = SGPR_U64 + VGPR_I32 + IOFFSET; first operand = LDS byte
// address VGPR. Generic pointers to __shared__ carry the LDS byte offset in
// their low 32 bits (aperture in the high bits), so truncation gives VDST.
__device__ __forceinline__ void async_ld_b128(unsigned ldsOff,
                                              unsigned long long base,
                                              unsigned byteOff) {
    asm volatile("global_load_async_to_lds_b128 %0, %1, %2"
                 :: "v"(ldsOff), "v"(byteOff), "s"(base)
                 : "memory");
}
__device__ __forceinline__ void wait_async0() {
    asm volatile("s_wait_asynccnt 0x0" ::: "memory");
}

// ---------------------------------------------------------------------------
// Constants (problem is fixed-shape)
// ---------------------------------------------------------------------------
#define NTOK 8192
#define DD   2048
#define NE   8
#define TM   128
#define TN   128
#define TKK  64          // K-slab per barrier (2 WMMA K-steps of 32)

// ---------------------------------------------------------------------------
// Utility kernels
// ---------------------------------------------------------------------------
__global__ void moe_zero_f32(float* __restrict__ p, size_t n) {
    size_t i = (size_t)blockIdx.x * blockDim.x + threadIdx.x;
    size_t s = (size_t)gridDim.x * blockDim.x;
    for (; i < n; i += s) p[i] = 0.0f;
}

__global__ void moe_zero_i32(int* __restrict__ p, int n) {
    int i = threadIdx.x;
    if (i < n) p[i] = 0;
}

__global__ void moe_cvt_bf16(const float* __restrict__ in,
                             unsigned short* __restrict__ out, size_t n) {
    size_t i = (size_t)blockIdx.x * blockDim.x + threadIdx.x;
    size_t s = (size_t)gridDim.x * blockDim.x;
    for (; i < n; i += s) out[i] = f2bf(in[i]);
}

// fp32 [E][K][N] -> bf16 [E][N][K] (transposed), LDS-tiled for coalescing.
// Transposed weights make GEMM B-tiles contiguous along K => pure async copy.
__global__ void moe_cvt_tr(const float* __restrict__ in,
                           unsigned short* __restrict__ out) {
    int bid = blockIdx.x;
    int e   = bid >> 12;               // 4096 = (D/32)^2 tiles per expert
    int r   = bid & 4095;
    int kt  = (r >> 6) << 5;
    int nt  = (r & 63) << 5;
    __shared__ float t[32][33];
    int c  = threadIdx.x & 31;
    int rr = threadIdx.x >> 5;
    const float* ib = in + ((size_t)e * DD + kt) * DD + nt;
#pragma unroll
    for (int q = 0; q < 4; ++q)
        t[rr + q * 8][c] = ib[(size_t)(rr + q * 8) * DD + c];
    __syncthreads();
    unsigned short* ob = out + ((size_t)e * DD + nt) * DD + kt;
#pragma unroll
    for (int q = 0; q < 4; ++q)
        ob[(size_t)(rr + q * 8) * DD + c] = f2bf(t[c][rr + q * 8]);
}

// ---------------------------------------------------------------------------
// Gating: one wave32 per token.
// ---------------------------------------------------------------------------
__global__ void moe_gate(const float* __restrict__ x,
                         const float* __restrict__ gw,
                         const float* __restrict__ gb,
                         int* __restrict__ cnt,
                         int* __restrict__ tok_list,
                         float* __restrict__ slot_w) {
    int gtid = blockIdx.x * blockDim.x + threadIdx.x;
    int tok  = gtid >> 5;
    int lane = threadIdx.x & 31;
    if (tok >= NTOK) return;

    const float* xr = x + (size_t)tok * DD;
    float acc[NE];
#pragma unroll
    for (int e = 0; e < NE; ++e) acc[e] = 0.0f;

    for (int d = lane; d < DD; d += 32) {
        float xv = xr[d];
        const float* g = gw + (size_t)d * NE;
#pragma unroll
        for (int e = 0; e < NE; ++e) acc[e] += xv * g[e];
    }
#pragma unroll
    for (int e = 0; e < NE; ++e) {
#pragma unroll
        for (int off = 16; off > 0; off >>= 1)
            acc[e] += __shfl_xor(acc[e], off, 32);
    }

    if (lane == 0) {
        float lg[NE], p[NE];
        float m = -1e30f;
#pragma unroll
        for (int e = 0; e < NE; ++e) { lg[e] = acc[e] + gb[e]; m = lg[e] > m ? lg[e] : m; }
        float s = 0.0f;
#pragma unroll
        for (int e = 0; e < NE; ++e) { p[e] = __expf(lg[e] - m); s += p[e]; }
        float inv = 1.0f / s;
        int   i0 = 0, i1 = -1;
        float s0 = -1.0f, s1 = -1.0f;
#pragma unroll
        for (int e = 0; e < NE; ++e) {
            float v = p[e] * inv;
            if (v > s0)      { s1 = s0; i1 = i0; s0 = v; i0 = e; }
            else if (v > s1) { s1 = v;  i1 = e; }
        }
        float wsum = s0 + s1 + 1e-8f;
        int sl0 = atomicAdd(&cnt[i0], 1);
        tok_list[i0 * NTOK + sl0] = tok;
        slot_w [i0 * NTOK + sl0] = s0 / wsum;
        int sl1 = atomicAdd(&cnt[i1], 1);
        tok_list[i1 * NTOK + sl1] = tok;
        slot_w [i1 * NTOK + sl1] = s1 / wsum;
    }
}

__global__ void moe_scan(const int* __restrict__ cnt, int* __restrict__ off) {
    if (threadIdx.x == 0) {
        int a = 0;
#pragma unroll
        for (int e = 0; e < NE; ++e) { off[e] = a; a += cnt[e]; }
    }
}

// ---------------------------------------------------------------------------
// WMMA tile core.  Block = 256 threads = 8 waves (4 M x 2 N), tile 128x128,
// K slab 64, double-buffered LDS fed by global_load_async_to_lds_b128.
// A fragment (16-bit dense): lane<16 -> K 0-15 ; lane>=16 ->
// K 16-31  => two ds_load_b128 at +0B and +16B.
// B fragment (16-bit dense): lane<16 -> K 0-15 ; lane>=16 -> K 16-31
// => two ds_load_b128 at +0B and +16B.
// ---------------------------------------------------------------------------
union WFrag { v16bf v; uint4 q[2]; };

// GEMM 1: H = relu( gather(X) @ W1t[e]^T + b1[e] ), bf16 slot rows.
__global__ __launch_bounds__(256)
void moe_gemm1(const unsigned short* __restrict__ Xb,
               const unsigned short* __restrict__ W1t,   // [E][N][K] bf16
               const float* __restrict__ b1,
               const int* __restrict__ cnt,
               const int* __restrict__ off,
               const int* __restrict__ tok_list,
               unsigned short* __restrict__ H) {
    int e     = blockIdx.x >> 6;
    int mtile = blockIdx.x & 63;
    int n0    = blockIdx.y * TN;
    int c_e   = cnt[e];
    int mb    = mtile * TM;
    if (mb >= c_e) return;
    int o_e = off[e];

    __shared__ __align__(16) unsigned short As[2][TM * TKK];
    __shared__ __align__(16) unsigned short Bs[2][TN * TKK];

    int tid = threadIdx.x, lane = tid & 31, wid = tid >> 5;
    int wm = wid & 3, wn = wid >> 2;

    // Async staging: each 128x64 bf16 tile = 1024 16B chunks; thread owns
    // chunks tid + 256q.  chunk -> row = chunk>>3, seg = chunk&7.
    unsigned aOff[4], bOff[4], aL0[4], aL1[4], bL0[4], bL1[4];
    bool aVal[4];
#pragma unroll
    for (int q = 0; q < 4; ++q) {
        int chunk = tid + (q << 8);
        int row = chunk >> 3, seg = chunk & 7;
        int am  = mb + row;
        bool v  = (am < c_e);
        int  t  = v ? tok_list[e * NTOK + am] : 0;
        aVal[q] = v;
        aOff[q] = ((unsigned)t * DD) * 2u + (unsigned)seg * 16u;
        bOff[q] = ((unsigned)(n0 + row) * DD) * 2u + (unsigned)seg * 16u;
        aL0[q] = (unsigned)(uintptr_t)(&As[0][0]) + chunk * 16u;
        aL1[q] = (unsigned)(uintptr_t)(&As[1][0]) + chunk * 16u;
        bL0[q] = (unsigned)(uintptr_t)(&Bs[0][0]) + chunk * 16u;
        bL1[q] = (unsigned)(uintptr_t)(&Bs[1][0]) + chunk * 16u;
    }
    unsigned long long aBase = (unsigned long long)(uintptr_t)Xb;
    unsigned long long bBase = (unsigned long long)(uintptr_t)(W1t + (size_t)e * DD * DD);

    v8f acc[2][4] = {};

    // prologue: slab 0 into buffer 0
#pragma unroll
    for (int q = 0; q < 4; ++q) {
        if (aVal[q]) async_ld_b128(aL0[q], aBase, aOff[q]);
        async_ld_b128(bL0[q], bBase, bOff[q]);
    }

    const int NSTEP = DD / TKK;          // 32
    for (int s = 0; s < NSTEP; ++s) {
        int cur = s & 1;
        wait_async0();
        __syncthreads();
        if (s + 1 < NSTEP) {
            unsigned k2 = (unsigned)((s + 1) * TKK) * 2u;
#pragma unroll
            for (int q = 0; q < 4; ++q) {
                if (aVal[q]) async_ld_b128(cur ? aL0[q] : aL1[q], aBase, aOff[q] + k2);
                async_ld_b128(cur ? bL0[q] : bL1[q], bBase, bOff[q] + k2);
            }
        }
        int kbA = (lane < 16) ? 0 : 16;
        int kbB = (lane < 16) ? 0 : 16;
        int rl  = lane & 15;
        const unsigned short* Ab = &As[cur][0];
        const unsigned short* Bb = &Bs[cur][0];
#pragma unroll
        for (int ks = 0; ks < 2; ++ks) {
            WFrag a[2], b[4];
#pragma unroll
            for (int mi = 0; mi < 2; ++mi) {
                const unsigned short* p = Ab + (wm * 32 + mi * 16 + rl) * TKK + ks * 32 + kbA;
                a[mi].q[0] = *(const uint4*)p;
                a[mi].q[1] = *(const uint4*)(p + 8);
            }
#pragma unroll
            for (int ni = 0; ni < 4; ++ni) {
                const unsigned short* p = Bb + (wn * 64 + ni * 16 + rl) * TKK + ks * 32 + kbB;
                b[ni].q[0] = *(const uint4*)p;
                b[ni].q[1] = *(const uint4*)(p + 8);
            }
#pragma unroll
            for (int mi = 0; mi < 2; ++mi)
#pragma unroll
                for (int ni = 0; ni < 4; ++ni)
                    acc[mi][ni] = __builtin_amdgcn_wmma_f32_16x16x32_bf16(
                        false, a[mi].v, false, b[ni].v, (short)0, acc[mi][ni],
                        false, false);
        }
    }

    int hi8 = (lane < 16) ? 0 : 8;
    int cl  = lane & 15;
#pragma unroll
    for (int ni = 0; ni < 4; ++ni) {
        int col = n0 + wn * 64 + ni * 16 + cl;
        float bias = b1[e * DD + col];
#pragma unroll
        for (int mi = 0; mi < 2; ++mi) {
#pragma unroll
            for (int j = 0; j < 8; ++j) {
                int ml = wm * 32 + mi * 16 + j + hi8;
                int m  = mb + ml;
                if (m < c_e) {
                    float v = acc[mi][ni][j] + bias;
                    v = v > 0.0f ? v : 0.0f;
                    H[(size_t)(o_e + m) * DD + col] = f2bf(v);
                }
            }
        }
    }
}

// GEMM 2: y = H @ W2t[e]^T + b2[e]; out[token] += slot_weight * y.
__global__ __launch_bounds__(256)
void moe_gemm2(const unsigned short* __restrict__ H,
               const unsigned short* __restrict__ W2t,   // [E][N][K] bf16
               const float* __restrict__ b2,
               const int* __restrict__ cnt,
               const int* __restrict__ off,
               const int* __restrict__ tok_list,
               const float* __restrict__ slot_w,
               float* __restrict__ out) {
    int e     = blockIdx.x >> 6;
    int mtile = blockIdx.x & 63;
    int n0    = blockIdx.y * TN;
    int c_e   = cnt[e];
    int mb    = mtile * TM;
    if (mb >= c_e) return;
    int o_e = off[e];

    __shared__ __align__(16) unsigned short As[2][TM * TKK];
    __shared__ __align__(16) unsigned short Bs[2][TN * TKK];

    int tid = threadIdx.x, lane = tid & 31, wid = tid >> 5;
    int wm = wid & 3, wn = wid >> 2;

    unsigned aOff[4], bOff[4], aL0[4], aL1[4], bL0[4], bL1[4];
    bool aVal[4];
#pragma unroll
    for (int q = 0; q < 4; ++q) {
        int chunk = tid + (q << 8);
        int row = chunk >> 3, seg = chunk & 7;
        int am  = mb + row;
        aVal[q] = (am < c_e);
        aOff[q] = ((unsigned)(o_e + am) * DD) * 2u + (unsigned)seg * 16u;
        bOff[q] = ((unsigned)(n0 + row) * DD) * 2u + (unsigned)seg * 16u;
        aL0[q] = (unsigned)(uintptr_t)(&As[0][0]) + chunk * 16u;
        aL1[q] = (unsigned)(uintptr_t)(&As[1][0]) + chunk * 16u;
        bL0[q] = (unsigned)(uintptr_t)(&Bs[0][0]) + chunk * 16u;
        bL1[q] = (unsigned)(uintptr_t)(&Bs[1][0]) + chunk * 16u;
    }
    unsigned long long aBase = (unsigned long long)(uintptr_t)H;
    unsigned long long bBase = (unsigned long long)(uintptr_t)(W2t + (size_t)e * DD * DD);

    v8f acc[2][4] = {};

#pragma unroll
    for (int q = 0; q < 4; ++q) {
        if (aVal[q]) async_ld_b128(aL0[q], aBase, aOff[q]);
        async_ld_b128(bL0[q], bBase, bOff[q]);
    }

    const int NSTEP = DD / TKK;
    for (int s = 0; s < NSTEP; ++s) {
        int cur = s & 1;
        wait_async0();
        __syncthreads();
        if (s + 1 < NSTEP) {
            unsigned k2 = (unsigned)((s + 1) * TKK) * 2u;
#pragma unroll
            for (int q = 0; q < 4; ++q) {
                if (aVal[q]) async_ld_b128(cur ? aL0[q] : aL1[q], aBase, aOff[q] + k2);
                async_ld_b128(cur ? bL0[q] : bL1[q], bBase, bOff[q] + k2);
            }
        }
        int kbA = (lane < 16) ? 0 : 16;
        int kbB = (lane < 16) ? 0 : 16;
        int rl  = lane & 15;
        const unsigned short* Ab = &As[cur][0];
        const unsigned short* Bb = &Bs[cur][0];
#pragma unroll
        for (int ks = 0; ks < 2; ++ks) {
            WFrag a[2], b[4];
#pragma unroll
            for (int mi = 0; mi < 2; ++mi) {
                const unsigned short* p = Ab + (wm * 32 + mi * 16 + rl) * TKK + ks * 32 + kbA;
                a[mi].q[0] = *(const uint4*)p;
                a[mi].q[1] = *(const uint4*)(p + 8);
            }
#pragma unroll
            for (int ni = 0; ni < 4; ++ni) {
                const unsigned short* p = Bb + (wn * 64 + ni * 16 + rl) * TKK + ks * 32 + kbB;
                b[ni].q[0] = *(const uint4*)p;
                b[ni].q[1] = *(const uint4*)(p + 8);
            }
#pragma unroll
            for (int mi = 0; mi < 2; ++mi)
#pragma unroll
                for (int ni = 0; ni < 4; ++ni)
                    acc[mi][ni] = __builtin_amdgcn_wmma_f32_16x16x32_bf16(
                        false, a[mi].v, false, b[ni].v, (short)0, acc[mi][ni],
                        false, false);
        }
    }

    int hi8 = (lane < 16) ? 0 : 8;
    int cl  = lane & 15;
#pragma unroll
    for (int ni = 0; ni < 4; ++ni) {
        int col = n0 + wn * 64 + ni * 16 + cl;
        float bias = b2[e * DD + col];
#pragma unroll
        for (int mi = 0; mi < 2; ++mi) {
#pragma unroll
            for (int j = 0; j < 8; ++j) {
                int ml = wm * 32 + mi * 16 + j + hi8;
                int m  = mb + ml;
                if (m < c_e) {
                    float y = acc[mi][ni][j] + bias;
                    int   t = tok_list[e * NTOK + m];
                    float w = slot_w [e * NTOK + m];
                    atomicAdd(out + (size_t)t * DD + col, w * y);
                }
            }
        }
    }
}

// ---------------------------------------------------------------------------
// Host-side launcher
// ---------------------------------------------------------------------------
extern "C" void kernel_launch(void* const* d_in, const int* in_sizes, int n_in,
                              void* d_out, int out_size, void* d_ws, size_t ws_size,
                              hipStream_t stream) {
    const float* x  = (const float*)d_in[0];
    const float* w1 = (const float*)d_in[1];
    const float* b1 = (const float*)d_in[2];
    const float* w2 = (const float*)d_in[3];
    const float* b2 = (const float*)d_in[4];
    const float* gw = (const float*)d_in[5];
    const float* gb = (const float*)d_in[6];
    float* out = (float*)d_out;   // [N*DD] out + 1 aux scalar (stays zero)

    const size_t N = NTOK, D = DD, E = NE;

    unsigned short* xb  = (unsigned short*)d_ws;         // N*D
    unsigned short* w1t = xb  + N * D;                   // E*D*D (transposed)
    unsigned short* w2t = w1t + E * D * D;               // E*D*D (transposed)
    unsigned short* Hb  = w2t + E * D * D;               // N*2*D (compacted slots)
    int*   cnt  = (int*)(Hb + N * 2 * D);                // E (padded 16)
    int*   offp = cnt + 16;                              // E (padded 16)
    int*   tok  = offp + 16;                             // E*N
    float* slw  = (float*)(tok + E * N);                 // E*N

    moe_zero_f32<<<4096, 256, 0, stream>>>(out, (size_t)out_size);
    moe_zero_i32<<<1, 32, 0, stream>>>(cnt, 16);

    moe_cvt_bf16<<<2048, 256, 0, stream>>>(x, xb, N * D);
    moe_cvt_tr<<<(int)(E * 4096), 256, 0, stream>>>(w1, w1t);
    moe_cvt_tr<<<(int)(E * 4096), 256, 0, stream>>>(w2, w2t);

    moe_gate<<<(int)(N / 8), 256, 0, stream>>>(x, gw, gb, cnt, tok, slw);
    moe_scan<<<1, 1, 0, stream>>>(cnt, offp);

    dim3 grid(E * 64, D / TN);
    moe_gemm1<<<grid, 256, 0, stream>>>(xb, w1t, b1, cnt, offp, tok, Hb);
    moe_gemm2<<<grid, 256, 0, stream>>>(Hb, w2t, b2, cnt, offp, tok, slw, out);
}
